// SelfAttention_29566554866179
// MI455X (gfx1250) — matrix operand order repaired and checked by reading, not compiled, not run on hardware
//
#include <hip/hip_runtime.h>
#include <hip/hip_bf16.h>

// Problem constants (fixed by the reference setup)
#define B_ 4
#define T_ 2048
#define C_ 1024
#define H_ 16
#define HD_ 64

// Toggle for the Tensor Data Mover staging path in the attention kernel.
#define USE_TDM 1

typedef __attribute__((ext_vector_type(16))) _Float16 v16h;
typedef __attribute__((ext_vector_type(8)))  _Float16 v8h;
typedef __attribute__((ext_vector_type(8)))  float    v8f;
typedef __attribute__((ext_vector_type(4)))  unsigned int v4u;
typedef __attribute__((ext_vector_type(8)))  unsigned int v8u;

static __device__ __forceinline__ v16h cat8(v8h a, v8h b) {
    return __builtin_shufflevector(a, b, 0,1,2,3,4,5,6,7,8,9,10,11,12,13,14,15);
}

static __device__ __forceinline__ v8f wmma_f16(v16h a, v16h b, v8f c) {
    return __builtin_amdgcn_wmma_f32_16x16x32_f16(
        /*neg_a=*/false, a, /*neg_b=*/false, b,
        /*c_mod=*/(short)0, c, /*reuse_a=*/false, /*reuse_b=*/false);
}

static __device__ __forceinline__ void wait_tensorcnt0() {
#if __has_builtin(__builtin_amdgcn_s_wait_tensorcnt)
    __builtin_amdgcn_s_wait_tensorcnt(0);
#else
    asm volatile("s_wait_tensorcnt 0x0" ::: "memory");
#endif
}

// Issue a TDM 2D tile load: global (row-major, rowlen halfs, nrows rows,
// stride = rowlen) -> LDS at lds_byte_addr with +4 DWORD padding per
// 32-DWORD (128B) row, i.e. LDS row stride 144B = 72 halfs.
static __device__ __forceinline__ void tdm_load_tile_f16(unsigned int lds_byte_addr,
                                                         unsigned long long gaddr) {
    v4u g0;
    g0[0] = 1u;                                   // count=1, user descriptor
    g0[1] = lds_byte_addr;                        // lds_addr [63:32]
    g0[2] = (unsigned int)gaddr;                  // global_addr [95:64]
    g0[3] = ((unsigned int)(gaddr >> 32) & 0x01FFFFFFu) | (2u << 30); // [120:96], type=2

    v8u g1;
    g1[0] = (1u << 16)        // data_size = 1 -> 2 bytes
          | (1u << 20)        // pad_enable
          | (4u << 22)        // pad_interval code 4 -> every 32 DWORDs (128B)
          | (3u << 25);       // pad_amount  code 3 -> 4 DWORDs (16B)
    g1[1] = (64u << 16);      // tensor_dim0 = 64 (bits 79:48)
    g1[2] = (32u << 16);      // tensor_dim1 = 32 (bits 111:80)
    g1[3] = (64u << 16);      // tile_dim0 = 64 (bits 127:112)
    g1[4] = 32u;              // tile_dim1 = 32 (bits 143:128), tile_dim2 = 0
    g1[5] = 64u;              // tensor_dim0_stride = 64 (bits 207:160)
    g1[6] = 0u;
    g1[7] = 0u;

    v4u gz = {0u, 0u, 0u, 0u};                    // groups 2/3: unused dims

    asm volatile("tensor_load_to_lds %0, %1, %2, %3"
                 :
                 : "s"(g0), "s"(g1), "s"(gz), "s"(gz)
                 : "memory");
}

// ---------------------------------------------------------------------------
// Kernel 1: fp32 -> f16 conversion (weights)
// ---------------------------------------------------------------------------
__global__ __launch_bounds__(256) void cvt_f32_f16_kernel(const float* __restrict__ s,
                                                          _Float16* __restrict__ d, int n) {
    int i = blockIdx.x * blockDim.x + threadIdx.x;
    if (i < n) d[i] = (_Float16)s[i];
}

// ---------------------------------------------------------------------------
// Kernel 2: projection GEMM  Y[b,h,t,hd] = sum_c X[b,t,c] * W[d,c]  (d=h*HD+hd)
// ---------------------------------------------------------------------------
__global__ __launch_bounds__(256) void proj_gemm_kernel(const float* __restrict__ X,
                                                        const _Float16* __restrict__ W16,
                                                        _Float16* __restrict__ Yhead) {
    __shared__ __align__(16) _Float16 Xs[128 * 40];
    __shared__ __align__(16) _Float16 Wsh[64 * 40];

    const int tid  = threadIdx.x;
    const int bm   = blockIdx.y * 128;
    const int bn   = blockIdx.x * 64;
    const int wid  = tid >> 5;
    const int lane = tid & 31;
    const int l16  = lane & 15;
    const int hi   = lane >> 4;
    const int wm   = wid & 3;
    const int wn   = wid >> 2;
    const int b0   = hi * 8;
    const int bb   = hi * 16;

    v8f acc[2][2];
    #pragma unroll
    for (int i = 0; i < 2; ++i)
        #pragma unroll
        for (int j = 0; j < 2; ++j) acc[i][j] = (v8f){};

    for (int kk = 0; kk < C_; kk += 32) {
        {
            const int r  = tid >> 1;
            const int c0 = (tid & 1) * 16;
            const float* src = X + (size_t)(bm + r) * C_ + kk + c0;
            if (kk + 32 < C_) __builtin_prefetch(src + 32, 0, 1);
            v8h t0, t1;
            #pragma unroll
            for (int i = 0; i < 8; ++i) { t0[i] = (_Float16)src[i]; t1[i] = (_Float16)src[8 + i]; }
            *(v8h*)&Xs[r * 40 + c0]     = t0;
            *(v8h*)&Xs[r * 40 + c0 + 8] = t1;
        }
        {
            const int r  = tid >> 2;
            const int c0 = (tid & 3) * 8;
            *(v8h*)&Wsh[r * 40 + c0] = *(const v8h*)&W16[(size_t)(bn + r) * C_ + kk + c0];
        }
        __syncthreads();

        v16h afrag[2], bfrag[2];
        #pragma unroll
        for (int ms = 0; ms < 2; ++ms) {
            const int m = wm * 32 + ms * 16 + l16;
            afrag[ms] = cat8(*(const v8h*)&Xs[m * 40 + bb],
                             *(const v8h*)&Xs[m * 40 + bb + 8]);
        }
        #pragma unroll
        for (int ns = 0; ns < 2; ++ns) {
            const int n = wn * 32 + ns * 16 + l16;
            bfrag[ns] = cat8(*(const v8h*)&Wsh[n * 40 + bb],
                             *(const v8h*)&Wsh[n * 40 + bb + 8]);
        }
        #pragma unroll
        for (int ms = 0; ms < 2; ++ms)
            #pragma unroll
            for (int ns = 0; ns < 2; ++ns)
                acc[ms][ns] = wmma_f16(afrag[ms], bfrag[ns], acc[ms][ns]);
        __syncthreads();
    }

    #pragma unroll
    for (int ms = 0; ms < 2; ++ms)
        #pragma unroll
        for (int ns = 0; ns < 2; ++ns)
            #pragma unroll
            for (int j = 0; j < 8; ++j) {
                const int gm = bm + wm * 32 + ms * 16 + hi * 8 + j;
                const int gn = bn + wn * 32 + ns * 16 + l16;
                const int b  = gm / T_, t = gm % T_;
                const int h  = gn / HD_, hd = gn % HD_;
                Yhead[(((size_t)b * H_ + h) * T_ + t) * HD_ + hd] = (_Float16)acc[ms][ns][j];
            }
}

// ---------------------------------------------------------------------------
// Kernel 3: flash attention; K-block staged via Tensor Data Mover
// ---------------------------------------------------------------------------
__global__ __launch_bounds__(256) void attn_kernel(const _Float16* __restrict__ Qp,
                                                   const _Float16* __restrict__ Kp,
                                                   const _Float16* __restrict__ Vp,
                                                   _Float16* __restrict__ AO) {
    __shared__ __align__(16) _Float16 Kblk[32 * 72];     // [key][hd], stride 72 (TDM-padded)
    __shared__ __align__(16) _Float16 VblkT[64 * 40];    // [hd][key], stride 40
    __shared__ __align__(16) _Float16 Pbuf[8 * 16 * 32]; // per-wave P tile 16x32

    const int tid  = threadIdx.x;
    const int bh   = blockIdx.y;        // b*H + h
    const int b    = bh >> 4;
    const int h    = bh & 15;
    const int q0   = blockIdx.x * 128;
    const int wid  = tid >> 5;
    const int lane = tid & 31;
    const int l16  = lane & 15;
    const int hi   = lane >> 4;
    const int b0   = hi * 8;
    const int bb   = hi * 16;

    const _Float16* Qh = Qp + (size_t)bh * T_ * HD_;
    const _Float16* Kh = Kp + (size_t)bh * T_ * HD_;
    const _Float16* Vh = Vp + (size_t)bh * T_ * HD_;
    _Float16* Pw = &Pbuf[wid * 512];

    const int qrowbase = q0 + wid * 16;

    v16h qfrag[2];
    {
        const int qr = qrowbase + l16;
        #pragma unroll
        for (int c = 0; c < 2; ++c)
            qfrag[c] = cat8(*(const v8h*)&Qh[(size_t)qr * HD_ + c * 32 + bb],
                            *(const v8h*)&Qh[(size_t)qr * HD_ + c * 32 + bb + 8]);
    }

    float m[8], l[8];
    v8f O[4];
    #pragma unroll
    for (int j = 0; j < 8; ++j) { m[j] = -3.0e38f; l[j] = 0.0f; }
    #pragma unroll
    for (int t = 0; t < 4; ++t) O[t] = (v8f){};

#if USE_TDM
    const unsigned int kblk_lds = (unsigned int)(unsigned long long)(uintptr_t)&Kblk[0];
#endif

    const int nkb = q0 / 32 + 4;   // causal: keys up to q0+127
    for (int kb = 0; kb < nkb; ++kb) {
        __syncthreads();
#if USE_TDM
        // K block via Tensor Data Mover (wave 0 issues; TENSORcnt tracks it)
        if (wid == 0) {
            tdm_load_tile_f16(kblk_lds,
                              (unsigned long long)(uintptr_t)(Kh + (size_t)(kb * 32) * HD_));
        }
#else
        {
            const int r  = tid >> 3;
            const int c0 = (tid & 7) * 8;
            v8h kv = *(const v8h*)&Kh[(size_t)(kb * 32 + r) * HD_ + c0];
            *(v8h*)&Kblk[r * 72 + c0] = kv;
        }
#endif
        // V block staged transposed by all threads
        {
            const int r  = tid >> 3;          // 0..31 key row
            const int c0 = (tid & 7) * 8;     // hd offset
            v8h vv = *(const v8h*)&Vh[(size_t)(kb * 32 + r) * HD_ + c0];
            #pragma unroll
            for (int i = 0; i < 8; ++i) VblkT[(c0 + i) * 40 + r] = vv[i];
        }
#if USE_TDM
        if (wid == 0) wait_tensorcnt0();
#endif
        __syncthreads();

        // S = Q @ K^T
        v8f s[2];
        s[0] = (v8f){}; s[1] = (v8f){};
        #pragma unroll
        for (int nt = 0; nt < 2; ++nt) {
            const int krow = nt * 16 + l16;
            #pragma unroll
            for (int c = 0; c < 2; ++c) {
                v16h bfrag = cat8(*(const v8h*)&Kblk[krow * 72 + c * 32 + bb],
                                  *(const v8h*)&Kblk[krow * 72 + c * 32 + bb + 8]);
                s[nt] = wmma_f16(qfrag[c], bfrag, s[nt]);
            }
        }

        // online softmax (scale = 1/sqrt(B) = 0.5), causal mask, write P
        const int kidx0 = kb * 32 + l16;
        #pragma unroll
        for (int j = 0; j < 8; ++j) {
            const int qidx = qrowbase + hi * 8 + j;
            float v0 = s[0][j] * 0.5f;
            float v1 = s[1][j] * 0.5f;
            if (kidx0 > qidx)      v0 = -1.0e30f;
            if (kidx0 + 16 > qidx) v1 = -1.0e30f;
            float cur = fmaxf(v0, v1);
            #pragma unroll
            for (int o = 1; o < 16; o <<= 1) cur = fmaxf(cur, __shfl_xor(cur, o, 16));
            const float mnew = fmaxf(m[j], cur);
            const float p0 = __expf(v0 - mnew);
            const float p1 = __expf(v1 - mnew);
            float rs = p0 + p1;
            #pragma unroll
            for (int o = 1; o < 16; o <<= 1) rs += __shfl_xor(rs, o, 16);
            const float alpha = __expf(m[j] - mnew);
            l[j] = l[j] * alpha + rs;
            m[j] = mnew;
            #pragma unroll
            for (int t = 0; t < 4; ++t) O[t][j] *= alpha;
            const int prow = hi * 8 + j;
            Pw[prow * 32 + l16]      = (_Float16)p0;
            Pw[prow * 32 + 16 + l16] = (_Float16)p1;
        }
        __syncthreads();

        // O += P @ Vblk
        v16h pfrag = cat8(*(const v8h*)&Pw[l16 * 32 + bb],
                          *(const v8h*)&Pw[l16 * 32 + bb + 8]);
        #pragma unroll
        for (int t = 0; t < 4; ++t) {
            const int d = t * 16 + l16;
            v16h vfrag = cat8(*(const v8h*)&VblkT[d * 40 + bb],
                              *(const v8h*)&VblkT[d * 40 + bb + 8]);
            O[t] = wmma_f16(pfrag, vfrag, O[t]);
        }
    }

    #pragma unroll
    for (int t = 0; t < 4; ++t)
        #pragma unroll
        for (int j = 0; j < 8; ++j) {
            const int qidx = qrowbase + hi * 8 + j;
            const int col  = h * HD_ + t * 16 + l16;
            AO[((size_t)b * T_ + qidx) * C_ + col] = (_Float16)(O[t][j] / l[j]);
        }
}

// ---------------------------------------------------------------------------
// Kernel 4: output GEMM  out[m,n] = sum_c A[m,c] * Wo[n,c] + bo[n]   (fp32 out)
// ---------------------------------------------------------------------------
__global__ __launch_bounds__(256) void out_gemm_kernel(const _Float16* __restrict__ A,
                                                       const _Float16* __restrict__ W16,
                                                       const float* __restrict__ bias,
                                                       float* __restrict__ out) {
    __shared__ __align__(16) _Float16 Xs[128 * 40];
    __shared__ __align__(16) _Float16 Wsh[64 * 40];

    const int tid  = threadIdx.x;
    const int bm   = blockIdx.y * 128;
    const int bn   = blockIdx.x * 64;
    const int wid  = tid >> 5;
    const int lane = tid & 31;
    const int l16  = lane & 15;
    const int hi   = lane >> 4;
    const int wm   = wid & 3;
    const int wn   = wid >> 2;
    const int b0   = hi * 8;
    const int bb   = hi * 16;

    v8f acc[2][2];
    #pragma unroll
    for (int i = 0; i < 2; ++i)
        #pragma unroll
        for (int j = 0; j < 2; ++j) acc[i][j] = (v8f){};

    for (int kk = 0; kk < C_; kk += 32) {
        {
            const int r  = tid >> 1;
            const int c0 = (tid & 1) * 16;
            const _Float16* src = A + (size_t)(bm + r) * C_ + kk + c0;
            *(v8h*)&Xs[r * 40 + c0]     = *(const v8h*)(src);
            *(v8h*)&Xs[r * 40 + c0 + 8] = *(const v8h*)(src + 8);
        }
        {
            const int r  = tid >> 2;
            const int c0 = (tid & 3) * 8;
            *(v8h*)&Wsh[r * 40 + c0] = *(const v8h*)&W16[(size_t)(bn + r) * C_ + kk + c0];
        }
        __syncthreads();

        v16h afrag[2], bfrag[2];
        #pragma unroll
        for (int ms = 0; ms < 2; ++ms) {
            const int mm = wm * 32 + ms * 16 + l16;
            afrag[ms] = cat8(*(const v8h*)&Xs[mm * 40 + bb],
                             *(const v8h*)&Xs[mm * 40 + bb + 8]);
        }
        #pragma unroll
        for (int ns = 0; ns < 2; ++ns) {
            const int nn = wn * 32 + ns * 16 + l16;
            bfrag[ns] = cat8(*(const v8h*)&Wsh[nn * 40 + bb],
                             *(const v8h*)&Wsh[nn * 40 + bb + 8]);
        }
        #pragma unroll
        for (int ms = 0; ms < 2; ++ms)
            #pragma unroll
            for (int ns = 0; ns < 2; ++ns)
                acc[ms][ns] = wmma_f16(afrag[ms], bfrag[ns], acc[ms][ns]);
        __syncthreads();
    }

    #pragma unroll
    for (int ms = 0; ms < 2; ++ms)
        #pragma unroll
        for (int ns = 0; ns < 2; ++ns)
            #pragma unroll
            for (int j = 0; j < 8; ++j) {
                const int gm = bm + wm * 32 + ms * 16 + hi * 8 + j;
                const int gn = bn + wn * 32 + ns * 16 + l16;
                out[(size_t)gm * C_ + gn] = acc[ms][ns][j] + bias[gn];
            }
}

// ---------------------------------------------------------------------------
extern "C" void kernel_launch(void* const* d_in, const int* in_sizes, int n_in,
                              void* d_out, int out_size, void* d_ws, size_t ws_size,
                              hipStream_t stream) {
    (void)in_sizes; (void)n_in; (void)out_size; (void)ws_size;
    const float* v  = (const float*)d_in[0];
    const float* k  = (const float*)d_in[1];
    const float* q  = (const float*)d_in[2];
    // d_in[3] = mask (causal, implicit in kernel)
    const float* Wk = (const float*)d_in[4];
    const float* Wq = (const float*)d_in[5];
    const float* Wv = (const float*)d_in[6];
    const float* Wo = (const float*)d_in[7];
    const float* bo = (const float*)d_in[8];
    float* out = (float*)d_out;

    char* w = (char*)d_ws;
    const size_t WSZ = (size_t)C_ * C_ * sizeof(_Float16);       // 2 MB
    const size_t PSZ = (size_t)B_ * T_ * C_ * sizeof(_Float16);  // 16 MB
    _Float16* Wk16 = (_Float16*)(w);
    _Float16* Wq16 = (_Float16*)(w + WSZ);
    _Float16* Wv16 = (_Float16*)(w + 2 * WSZ);
    _Float16* Wo16 = (_Float16*)(w + 3 * WSZ);
    _Float16* Kp   = (_Float16*)(w + 4 * WSZ);
    _Float16* Qp   = (_Float16*)(w + 4 * WSZ + PSZ);
    _Float16* Vp   = (_Float16*)(w + 4 * WSZ + 2 * PSZ);
    _Float16* AO   = (_Float16*)(w + 4 * WSZ + 3 * PSZ);

    const int nW = C_ * C_;
    cvt_f32_f16_kernel<<<(nW + 255) / 256, 256, 0, stream>>>(Wk, Wk16, nW);
    cvt_f32_f16_kernel<<<(nW + 255) / 256, 256, 0, stream>>>(Wq, Wq16, nW);
    cvt_f32_f16_kernel<<<(nW + 255) / 256, 256, 0, stream>>>(Wv, Wv16, nW);
    cvt_f32_f16_kernel<<<(nW + 255) / 256, 256, 0, stream>>>(Wo, Wo16, nW);

    dim3 gg(C_ / 64, (B_ * T_) / 128);   // (16, 64)
    proj_gemm_kernel<<<gg, 256, 0, stream>>>(k, Wk16, Kp);
    proj_gemm_kernel<<<gg, 256, 0, stream>>>(q, Wq16, Qp);
    proj_gemm_kernel<<<gg, 256, 0, stream>>>(v, Wv16, Vp);

    dim3 ga(T_ / 128, B_ * H_);          // (16, 64)
    attn_kernel<<<ga, 256, 0, stream>>>(Qp, Kp, Vp, AO);

    out_gemm_kernel<<<gg, 256, 0, stream>>>(AO, Wo16, bo, out);
}
